// S4D_47193100648727
// MI455X (gfx1250) — matrix operand order repaired and checked by reading, not compiled, not run on hardware
//
#include <hip/hip_runtime.h>
#include <hip/hip_bf16.h>
#include <math.h>

// ---------------------------------------------------------------------------
// S4D fractional-kernel SSM, CDNA5 (gfx1250) WMMA implementation.
//   B=8, H=256, N=32, L=2048
// ---------------------------------------------------------------------------

typedef __attribute__((ext_vector_type(16))) __bf16 v16bf;
typedef __attribute__((ext_vector_type(8)))  __bf16 v8bf;
typedef __attribute__((ext_vector_type(8)))  float  v8f;

#define Bn   8
#define Hn   256
#define Nn   32
#define Ln   2048
#define KOFF 16            // front zero pad of reversed kernel
#define KLEN 2096          // used portion staged in LDS (multiple of 8)
#define KSTR 2112          // row stride of reversed-K in workspace
#define UPAD 64            // front zero pad of bf16 u rows
#define USTR (Ln + UPAD)   // 2112, row stride of padded bf16 u

#if __has_builtin(__builtin_amdgcn_global_load_async_to_lds_b128)
#define HAVE_ASYNC_LDS 1
// exact parameter types per hipcc diagnostic: v4i in AS(1) / AS(3)
typedef int v4i_ __attribute__((vector_size(16)));
typedef __attribute__((address_space(1))) v4i_* async_gptr;
typedef __attribute__((address_space(3))) v4i_* async_lptr;
#endif

__device__ __forceinline__ __bf16 bfraw(unsigned short s) {
    __bf16 b; __builtin_memcpy(&b, &s, 2); return b;
}
__device__ __forceinline__ __bf16 f2bf(float f) {
    union { float f; unsigned u; } v; v.f = f;
    unsigned u = v.u + 0x7fffu + ((v.u >> 16) & 1u);   // RNE truncation
    return bfraw((unsigned short)(u >> 16));
}

// 16-term Mittag-Leffler E_{1/4,1}(z) with |z| clamped to 50 (matches ref).
__device__ __forceinline__ void ml16(float zr, float zi, float& outr, float& outi) {
    const float RG[16] = {   // 1 / Gamma(1 + 0.25*k)
        1.00000000f, 1.10325866f, 1.12837917f, 1.08806532f,
        1.00000000f, 0.88261509f, 0.75225278f, 0.62175167f,
        0.50000000f, 0.39227291f, 0.30090111f, 0.22609165f,
        0.16666667f, 0.12069884f, 0.08597180f, 0.06029108f };
    float az2 = zr * zr + zi * zi;
    if (az2 > 2500.0f) { float s = 50.0f * rsqrtf(az2); zr *= s; zi *= s; }
    float tr = 1.0f, ti = 0.0f;
    float sr = RG[0], si = 0.0f;
#pragma unroll
    for (int k = 1; k < 16; ++k) {
        float nr = tr * zr - ti * zi;
        float ni = tr * zi + ti * zr;
        tr = nr; ti = ni;
        sr = fmaf(tr, RG[k], sr);
        si = fmaf(ti, RG[k], si);
    }
    outr = sr; outi = si;
}

// --------------------------- kernel 1: build K -----------------------------
// writes reversed, padded, bf16 kernel rows: krev[h][i], i in [0,KSTR)
__global__ void s4d_k_kernel(const float* __restrict__ log_dt,
                             const float* __restrict__ C,      // (H,N,2)
                             const float* __restrict__ logAre, // (H,N)
                             const float* __restrict__ Aim,    // (H,N)
                             __bf16* __restrict__ krev) {
    int i = blockIdx.x * blockDim.x + threadIdx.x;
    int h = blockIdx.y;
    if (i >= KSTR) return;
    float val = 0.0f;
    if (i >= KOFF && i < KOFF + Ln) {
        int j = (KOFF + Ln - 1) - i;              // original time index
        float dt = __expf(log_dt[h]);
        float t1 = sqrtf(sqrtf((float)(j + 1) * dt));     // ((j+1)dt)^0.25
        float t0 = (j > 0) ? sqrtf(sqrtf((float)j * dt)) : 0.0f;
        float acc = 0.0f;
        for (int n = 0; n < Nn; ++n) {
            float ar = -__expf(logAre[h * Nn + n]);
            float ai = Aim[h * Nn + n];
            float cr = C[(h * Nn + n) * 2 + 0];
            float ci = C[(h * Nn + n) * 2 + 1];
            float e1r, e1i, e0r, e0i;
            ml16(ar * t1, ai * t1, e1r, e1i);
            ml16(ar * t0, ai * t0, e0r, e0i);     // z=0 -> ML=1 (matches ref)
            float edr = e1r - e0r, edi = e1i - e0i;
            float ia = 1.0f / (ar * ar + ai * ai);
            float xr = (edr * ar + edi * ai) * ia;        // E*conj(A)/|A|^2
            float xi = (edi * ar - edr * ai) * ia;
            acc += cr * xr - ci * xi;                     // Re[C * (E/A)]
        }
        val = 2.0f * acc;
    }
    krev[(size_t)h * KSTR + i] = f2bf(val);
}

// ----------------- u f32 -> bf16 with 64-element front pad -----------------
// grid (33, B*H), block 64: column c = blockIdx.x*64+tid in [0,USTR)
__global__ void s4d_ucvt_kernel(const float* __restrict__ x,
                                __bf16* __restrict__ y) {
    int c = blockIdx.x * 64 + threadIdx.x;
    int r = blockIdx.y;
    float v = (c >= UPAD) ? x[(size_t)r * Ln + (c - UPAD)] : 0.0f;
    y[(size_t)r * USTR + c] = f2bf(v);
}

// --------------------------- flat f32 -> bf16 ------------------------------
__global__ void s4d_cvt_kernel(const float* __restrict__ x,
                               __bf16* __restrict__ y, int n) {
    int i = blockIdx.x * blockDim.x + threadIdx.x;
    if (i < n) y[i] = f2bf(x[i]);
}

// ------------------- kernel 2: causal conv via WMMA ------------------------
// A (16x32 bf16) = u rows; M rows 0-7 = batches of tile tp, rows 8-15 =
//                  batches of tile tp+1 (B-block depends only on diagonal d).
// B (32x16 bf16) = two stacked 16x16 Toeplitz blocks of K (from LDS).
// C (16x16 f32)  = two y tiles (N = 16 consecutive l each).
__global__ void __launch_bounds__(128)
s4d_conv_kernel(const __bf16* __restrict__ ubf,  // (B,H) x USTR, front-padded
                const float*  __restrict__ u,    // (B,H,L) f32
                const float*  __restrict__ Dv,   // (H)
                const __bf16* __restrict__ krev, // (H,KSTR)
                __bf16* __restrict__ yb) {       // (B,L,H) bf16
    __shared__ __bf16 kl[KLEN];
    const int h   = blockIdx.x >> 4;
    const int g   = blockIdx.x & 15;
    const int tid = threadIdx.x;

    // stage reversed K row for this h into LDS (4.1 KB) via async DMA path
#ifdef HAVE_ASYNC_LDS
    {
        const __bf16* src = krev + (size_t)h * KSTR;
        for (int i = tid; i < KLEN / 8; i += blockDim.x) {
            __builtin_amdgcn_global_load_async_to_lds_b128(
                (async_gptr)(src + i * 8),
                (async_lptr)((__attribute__((address_space(3))) __bf16*)kl + i * 8),
                0, 0);
        }
#if __has_builtin(__builtin_amdgcn_s_wait_asynccnt)
        __builtin_amdgcn_s_wait_asynccnt(0);
#else
        asm volatile("s_wait_asynccnt 0x0" ::: "memory");
#endif
    }
#else
    {
        const uint4* src = (const uint4*)(krev + (size_t)h * KSTR);
        uint4* dst = (uint4*)kl;
        for (int i = tid; i < KLEN / 8; i += blockDim.x) dst[i] = src[i];
    }
#endif
    __syncthreads();

    const int wave = tid >> 5;
    const int lane = tid & 31;
    const int tp   = __builtin_amdgcn_readfirstlane(g * 8 + wave * 2); // even
    const int hi   = lane >> 4;
    const int lo   = lane & 15;

    const int bm = lo & 7;                        // batch for this A row
    // per-lane constant offset: tile select (lo>>3) + K-run half (-16*hi)
    const __bf16* ubase = ubf + ((size_t)(bm * Hn + h) * USTR + UPAD)
                              + (16 * (lo >> 3) - 16 * hi);

    v8f acc = {};
    v16bf av;
    {   // software pipeline: preload A for d = 0
        const __bf16* ap = ubase + 16 * tp;
        v8bf r0 = *(const v8bf*)(ap);
        v8bf r1 = *(const v8bf*)(ap + 8);
#pragma unroll
        for (int i = 0; i < 8; ++i) { av[i] = r0[i]; av[i + 8] = r1[i]; }
    }

    for (int d = 0; d <= tp; d += 2) {
        // B fragment: Toeplitz gather from reversed K (ascending in e)
        const int kb = (KOFF + Ln - 1) - 16 * d - lo - 16 * hi;
        v16bf bv;
#pragma unroll
        for (int e = 0; e < 16; ++e) bv[e] = kl[kb + e];

        // preload next A fragment (pad region returns zeros / unused data)
        v16bf nx;
        {
            const __bf16* ap = ubase + 16 * (tp - d - 2);
            v8bf r0 = *(const v8bf*)(ap);
            v8bf r1 = *(const v8bf*)(ap + 8);
#pragma unroll
            for (int i = 0; i < 8; ++i) { nx[i] = r0[i]; nx[i + 8] = r1[i]; }
        }
        if (d + 6 <= tp)
            __builtin_prefetch((const void*)(ubase + 16 * (tp - d - 6)), 0, 3);

        acc = __builtin_amdgcn_wmma_f32_16x16x32_bf16(
            false, av, false, bv, (short)0, acc, false, false);
        av = nx;
    }

    // epilogue: M = v + 8*hi -> batch = v, tile = tp + hi; N = lo
    {
        const int   l  = 16 * (tp + hi) + lo;
        const float dh = Dv[h];
#pragma unroll
        for (int v = 0; v < 8; ++v) {
            float y  = acc[v] + u[((size_t)v * Hn + h) * Ln + l] * dh;
            float ge = 0.5f * y * (1.0f + erff(y * 0.70710678f));   // exact gelu
            yb[((size_t)v * Ln + l) * Hn + h] = f2bf(ge);
        }
    }
}

// -------------- kernel 3: 1x1 conv GEMM (W @ y) + GLU via WMMA -------------
__global__ void __launch_bounds__(128)
s4d_glu_kernel(const __bf16* __restrict__ wbf,  // (2H,H)
               const __bf16* __restrict__ yb,   // (B*L, H)
               const float*  __restrict__ bias, // (2H)
               float* __restrict__ out) {       // (B,H,L)
    const int tid  = threadIdx.x;
    const int lane = tid & 31;
    const int tile = __builtin_amdgcn_readfirstlane(blockIdx.x * 4 + (tid >> 5));
    const int ot   = tile & 15;                 // oc tile
    const int ct   = tile >> 4;                 // column tile over B*L
    const int hi   = lane >> 4;
    const int lo   = lane & 15;

    v8f acc_a = {}, acc_g = {};
    const __bf16* wa = wbf + (size_t)(ot * 16 + lo) * Hn;
    const __bf16* wg = wa + (size_t)Hn * Hn;
    const __bf16* yc = yb + (size_t)(ct * 16 + lo) * Hn;

#pragma unroll
    for (int kk = 0; kk < Hn / 32; ++kk) {
        const int h0 = kk * 32 + 8 * hi;        // A fragment runs
        v8bf a0 = *(const v8bf*)(wa + h0);
        v8bf a1 = *(const v8bf*)(wa + h0 + 16);
        v8bf g0 = *(const v8bf*)(wg + h0);
        v8bf g1 = *(const v8bf*)(wg + h0 + 16);
        const int hb = kk * 32 + 8 * hi;        // B fragment runs
        v8bf b0 = *(const v8bf*)(yc + hb);
        v8bf b1 = *(const v8bf*)(yc + hb + 16);
        v16bf avv, gvv, bvv;
#pragma unroll
        for (int i = 0; i < 8; ++i) {
            avv[i] = a0[i]; avv[i + 8] = a1[i];
            gvv[i] = g0[i]; gvv[i + 8] = g1[i];
            bvv[i] = b0[i]; bvv[i + 8] = b1[i];
        }
        acc_a = __builtin_amdgcn_wmma_f32_16x16x32_bf16(
            false, avv, false, bvv, (short)0, acc_a, false, false);
        acc_g = __builtin_amdgcn_wmma_f32_16x16x32_bf16(
            false, gvv, false, bvv, (short)0, acc_g, false, false);
    }

    const int c0 = ct * 16;
    const int b  = c0 >> 11;                    // / 2048
    const int l  = (c0 & 2047) + lo;            // N = lo
#pragma unroll
    for (int v = 0; v < 8; ++v) {
        const int oc = ot * 16 + v + 8 * hi;    // M = v + 8*hi
        float aa = acc_a[v] + bias[oc];
        float gg = acc_g[v] + bias[oc + Hn];
        float sg = 1.0f / (1.0f + __expf(-gg));
        out[((size_t)b * Hn + oc) * Ln + l] = aa * sg;
    }
}

// ---------------------------------------------------------------------------
extern "C" void kernel_launch(void* const* d_in, const int* in_sizes, int n_in,
                              void* d_out, int out_size, void* d_ws, size_t ws_size,
                              hipStream_t stream) {
    const float* u      = (const float*)d_in[0];
    const float* log_dt = (const float*)d_in[1];
    const float* C      = (const float*)d_in[2];
    const float* logAre = (const float*)d_in[3];
    const float* Aim    = (const float*)d_in[4];
    const float* Dv     = (const float*)d_in[5];
    const float* W      = (const float*)d_in[6];
    const float* bias   = (const float*)d_in[7];
    float* out = (float*)d_out;

    char* ws = (char*)d_ws;
    __bf16* krev = (__bf16*)ws;  ws += (size_t)Hn * KSTR * 2;          // 1.08 MB
    __bf16* ubf  = (__bf16*)ws;  ws += (size_t)Bn * Hn * USTR * 2;     // 8.65 MB
    __bf16* wbf  = (__bf16*)ws;  ws += (size_t)2 * Hn * Hn * 2;        // 0.26 MB
    __bf16* yb   = (__bf16*)ws;                                        // 8.39 MB

    // 1) fractional kernel K (reversed, padded, bf16)
    dim3 gk((KSTR + 255) / 256, Hn);
    s4d_k_kernel<<<gk, 256, 0, stream>>>(log_dt, C, logAre, Aim, krev);

    // 2) bf16 conversions: u (front-padded rows) and W
    dim3 gu(USTR / 64, Bn * Hn);
    s4d_ucvt_kernel<<<gu, 64, 0, stream>>>(u, ubf);
    const int nw = 2 * Hn * Hn;
    s4d_cvt_kernel<<<(nw + 255) / 256, 256, 0, stream>>>(W, wbf, nw);

    // 3) causal Toeplitz-WMMA convolution + D-skip + GELU -> y (B,L,H) bf16
    s4d_conv_kernel<<<Hn * 16, 128, 0, stream>>>(ubf, u, Dv, krev, yb);

    // 4) 1x1 conv GEMM + GLU -> out (B,H,L) f32
    s4d_glu_kernel<<<(16 * 1024) / 4, 128, 0, stream>>>(wbf, yb, bias, out);
}
